// GAT_9689446220604
// MI455X (gfx1250) — matrix operand order repaired and checked by reading, not compiled, not run on hardware
//
#include <hip/hip_runtime.h>
#include <hip/hip_bf16.h>

// ---------------------------------------------------------------------------
// Types for CDNA5 WMMA (wave32): D[16x16 f32] = A[16x32 bf16] * B[32x16 bf16] + C
// ---------------------------------------------------------------------------
typedef __attribute__((ext_vector_type(16))) __bf16 v16bf;
typedef __attribute__((ext_vector_type(8)))  float  v8f;

#define HD      16      // heads*featsPerHead = 2*8
#define HF      1600    // MLP hidden
#define NT_HID  100     // 1600 / 16 column tiles
#define KSTEPS  50      // 1600 / 32 K steps
#define MT      32      // node rows per workgroup (two WMMA row-blocks)
#define NWAVES  8       // 256 threads / wave32

// ---------------------------------------------------------------------------
// Helpers
// ---------------------------------------------------------------------------
__device__ __forceinline__ __bf16 f2bf(float f) {
  unsigned u = __builtin_bit_cast(unsigned, f);
  unsigned r = (u + 0x7FFFu + ((u >> 16) & 1u)) >> 16;   // round-to-nearest-even
  return __builtin_bit_cast(__bf16, (unsigned short)r);
}
__device__ __forceinline__ __bf16 bfzero() {
  return __builtin_bit_cast(__bf16, (unsigned short)0);
}
// Order-preserving float<->uint map so atomicMax(u32) implements float max.
__device__ __forceinline__ unsigned enc_f(float f) {
  unsigned u = __builtin_bit_cast(unsigned, f);
  return (u & 0x80000000u) ? ~u : (u | 0x80000000u);
}
__device__ __forceinline__ float dec_f(unsigned e) {
  unsigned u = (e & 0x80000000u) ? (e & 0x7FFFFFFFu) : ~e;
  return __builtin_bit_cast(float, u);
}

// A fragment (16x32 bf16): row M = lane%16;
// lanes 0-15 hold K = k0+0..15, lanes 16-31 hold K = k0+16..31
__device__ __forceinline__ v16bf load_a_frag(const __bf16* base, int stride, int k0, int lane) {
  int row = lane & 15;
  int hi  = lane >> 4;
  const __bf16* p = base + row * stride + k0 + hi * 16;
  v16bf a;
  __builtin_memcpy(&a, p, 32);
  return a;
}
// B fragment (32x16 bf16): col N = lane%16; lanes 0-15 K=k0..k0+15, lanes 16-31 K=k0+16..31.
// Weight stored transposed [N][K] so a lane's 16 K values are contiguous.
__device__ __forceinline__ v16bf load_b_frag(const __bf16* wt, int ncol0, int ks, int k0, int lane) {
  int ncol = ncol0 + (lane & 15);
  int hi   = lane >> 4;
  const __bf16* p = wt + (size_t)ncol * ks + k0 + hi * 16;
  v16bf b;
  __builtin_memcpy(&b, p, 32);
  return b;
}
__device__ __forceinline__ v8f wmma_bf16(v16bf a, v16bf b, v8f c) {
  return __builtin_amdgcn_wmma_f32_16x16x32_bf16(false, a, false, b, (short)0, c, false, false);
}

// ---------------------------------------------------------------------------
// GAT kernels (memory/atomic bound; negligible FLOPs)
// ---------------------------------------------------------------------------
__global__ void gat_proj(const float* __restrict__ x, int fin,
                         const float* __restrict__ Ws, const float* __restrict__ Wd,
                         const float* __restrict__ Wr,
                         float* __restrict__ fs, float* __restrict__ fd,
                         float* __restrict__ r, int n) {
  int idx  = blockIdx.x * 256 + threadIdx.x;
  int node = idx >> 4, j = idx & 15;
  if (node >= n) return;
  float as = 0.f, ad = 0.f, ar = 0.f;
  for (int i = 0; i < fin; ++i) {
    float xv = x[(size_t)node * fin + i];
    as += xv * Ws[i * HD + j];
    ad += xv * Wd[i * HD + j];
    if (Wr) ar += xv * Wr[i * HD + j];
  }
  fs[idx] = as;
  fd[idx] = ad;
  if (Wr) r[idx] = ar;
}

__global__ void gat_edge_score(const int* __restrict__ src, const int* __restrict__ dst,
                               const float* __restrict__ fs, const float* __restrict__ fd,
                               const float* __restrict__ attn,
                               float* __restrict__ sbuf, unsigned* __restrict__ menc, int ne) {
  int e = blockIdx.x * 256 + threadIdx.x;
  if (e >= ne) return;
  int s = src[e], d = dst[e];
  for (int h = 0; h < 2; ++h) {
    float sc = 0.f;
    #pragma unroll
    for (int k = 0; k < 8; ++k) {
      float v = fs[(size_t)s * HD + h * 8 + k] + fd[(size_t)d * HD + h * 8 + k];
      v = v > 0.f ? v : 0.2f * v;                 // GATv2 leaky_relu
      sc += v * attn[h * 8 + k];
    }
    sbuf[(size_t)e * 2 + h] = sc;
    atomicMax(&menc[d * 2 + h], enc_f(sc));       // segment max
  }
}

__global__ void gat_edge_exp(const int* __restrict__ dst, float* __restrict__ sbuf,
                             const unsigned* __restrict__ menc, float* __restrict__ z, int ne2) {
  int i = blockIdx.x * 256 + threadIdx.x;
  if (i >= ne2) return;
  int e = i >> 1, h = i & 1;
  int d = dst[e];
  float m = dec_f(menc[d * 2 + h]);
  float p = __expf(sbuf[i] - m);
  sbuf[i] = p;
  atomicAdd(&z[d * 2 + h], p);                    // segment sum (partition)
}

__global__ void gat_edge_agg(const int* __restrict__ src, const int* __restrict__ dst,
                             const float* __restrict__ sbuf, const float* __restrict__ z,
                             const float* __restrict__ fs, float* __restrict__ acc, int ne2) {
  int i = blockIdx.x * 256 + threadIdx.x;
  if (i >= ne2) return;
  int e = i >> 1, h = i & 1;
  int s = src[e], d = dst[e];
  float alpha = sbuf[i] / z[d * 2 + h];
  #pragma unroll
  for (int k = 0; k < 8; ++k)
    atomicAdd(&acc[(size_t)d * HD + h * 8 + k], alpha * fs[(size_t)s * HD + h * 8 + k]);
}

__global__ void gat_finalize(const float* __restrict__ acc, const float* __restrict__ resid,
                             const float* __restrict__ bias, float* __restrict__ outn, int n) {
  int idx = blockIdx.x * 256 + threadIdx.x;
  int node = idx >> 4, j = idx & 15;
  if (node >= n) return;
  float v = acc[idx] + resid[idx] + bias[j];
  outn[idx] = v > 0.f ? v : (__expf(v) - 1.f);    // elu
}

// Concatenate [h_att1(16) | h_def2(16) | x(8) | pad(24)] as bf16, K padded to 64.
__global__ void pack_conc(const float* __restrict__ hA, const float* __restrict__ hD2,
                          const float* __restrict__ x, __bf16* __restrict__ conc, int n) {
  int idx = blockIdx.x * 256 + threadIdx.x;
  int node = idx >> 6, c = idx & 63;
  if (node >= n) return;
  float v = 0.f;
  if (c < 16)       v = hA[(size_t)node * HD + c];
  else if (c < 32)  v = hD2[(size_t)node * HD + (c - 16)];
  else if (c < 40)  v = x[(size_t)node * 8 + (c - 32)];
  conc[idx] = f2bf(v);
}

// ---------------------------------------------------------------------------
// Weight conversion: fp32 row-major [K][N] -> bf16 transposed [N][K] (+pad)
// ---------------------------------------------------------------------------
__global__ void conv_w1(const float* __restrict__ w, __bf16* __restrict__ wt) {
  int idx = blockIdx.x * 256 + threadIdx.x;          // [1600][64]
  if (idx >= HF * 64) return;
  int nn = idx >> 6, k = idx & 63;
  wt[idx] = (k < 40) ? f2bf(w[(size_t)k * HF + nn]) : bfzero();
}
__global__ void conv_w2(const float* __restrict__ w, __bf16* __restrict__ wt) {
  int idx = blockIdx.x * 256 + threadIdx.x;          // [1600][1600]
  if (idx >= HF * HF) return;
  int nn = idx / HF, k = idx % HF;
  wt[idx] = f2bf(w[(size_t)k * HF + nn]);
}
__global__ void conv_w3(const float* __restrict__ w, __bf16* __restrict__ wt) {
  int idx = blockIdx.x * 256 + threadIdx.x;          // [48][1600]
  if (idx >= 48 * HF) return;
  int nn = idx / HF, k = idx % HF;
  wt[idx] = (nn < 40) ? f2bf(w[(size_t)k * 40 + nn]) : bfzero();
}

// ---------------------------------------------------------------------------
// Fused MLP head: 40 -> 1600 -> 1600 -> 40 -> 1 per 32-row node tile.
// Two 16-row WMMA blocks share each B fragment (halves L2 weight traffic);
// all 32x1600 intermediates live in LDS as bf16 (215 KB/WGP -- CDNA5 only).
// ---------------------------------------------------------------------------
__global__ void __launch_bounds__(256)
mlp_fused(const __bf16* __restrict__ conc,  // [N][64]
          const __bf16* __restrict__ w1t,   // [1600][64]
          const float*  __restrict__ b1,
          const __bf16* __restrict__ w2t,   // [1600][1600]
          const float*  __restrict__ b2,
          const __bf16* __restrict__ w3t,   // [48][1600]
          const float*  __restrict__ b3,    // [40]
          const float*  __restrict__ w4,    // [40]
          const float*  __restrict__ b4,    // [1]
          float* __restrict__ out, int n) {
  __shared__ __bf16 sA [MT * 64];       //   4 KB  input tile
  __shared__ __bf16 sH1[MT * HF];       // 100 KB  layer-1 activations
  __shared__ __bf16 sH2[MT * HF];       // 100 KB  layer-2 activations
  __shared__ float  sH3[MT * 48];       //   6 KB  layer-3 activations

  const int tid  = threadIdx.x;
  const int wave = tid >> 5;
  const int lane = tid & 31;
  const int row0 = blockIdx.x * MT;
  const int hi   = lane >> 4;

  // Stage input tile (zero-pad rows past N).
  for (int i = tid; i < MT * 64; i += 256) {
    int r = i >> 6, c = i & 63;
    int gr = row0 + r;
    sA[i] = (gr < n) ? conc[(size_t)gr * 64 + c] : bfzero();
  }
  // Warm L2 path for the big weight.
  __builtin_prefetch(w2t + (size_t)tid * 64, 0, 1);
  __syncthreads();

  // ---- Layer 1: [32x64] @ [64x1600] ----
  for (int nt = wave; nt < NT_HID; nt += NWAVES) {
    v8f c0 = {}, c1 = {};
    #pragma unroll
    for (int kt = 0; kt < 2; ++kt) {
      v16bf b  = load_b_frag(w1t, nt * 16, 64, kt * 32, lane);
      v16bf a0 = load_a_frag(sA,           64, kt * 32, lane);
      v16bf a1 = load_a_frag(sA + 16 * 64, 64, kt * 32, lane);
      c0 = wmma_bf16(a0, b, c0);
      c1 = wmma_bf16(a1, b, c1);
    }
    int ncol = nt * 16 + (lane & 15);
    float bias = b1[ncol];
    #pragma unroll
    for (int i = 0; i < 8; ++i) {
      float v = c0[i] + bias;
      v = v > 0.f ? v : 0.01f * v;
      sH1[(i + hi * 8) * HF + ncol] = f2bf(v);
      float w = c1[i] + bias;
      w = w > 0.f ? w : 0.01f * w;
      sH1[(16 + i + hi * 8) * HF + ncol] = f2bf(w);
    }
  }
  __syncthreads();

  // ---- Layer 2: [32x1600] @ [1600x1600] (dominant GEMM) ----
  for (int nt = wave; nt < NT_HID; nt += NWAVES) {
    v8f c0 = {}, c1 = {};
    #pragma unroll 2
    for (int kt = 0; kt < KSTEPS; ++kt) {
      v16bf b  = load_b_frag(w2t, nt * 16, HF, kt * 32, lane);
      v16bf a0 = load_a_frag(sH1,           HF, kt * 32, lane);
      v16bf a1 = load_a_frag(sH1 + 16 * HF, HF, kt * 32, lane);
      c0 = wmma_bf16(a0, b, c0);
      c1 = wmma_bf16(a1, b, c1);
    }
    int ncol = nt * 16 + (lane & 15);
    float bias = b2[ncol];
    #pragma unroll
    for (int i = 0; i < 8; ++i) {
      float v = c0[i] + bias;
      v = v > 0.f ? v : 0.01f * v;
      sH2[(i + hi * 8) * HF + ncol] = f2bf(v);
      float w = c1[i] + bias;
      w = w > 0.f ? w : 0.01f * w;
      sH2[(16 + i + hi * 8) * HF + ncol] = f2bf(w);
    }
  }
  __syncthreads();

  // ---- Layer 3: [32x1600] @ [1600x48(40)] ----
  for (int nt = wave; nt < 3; nt += NWAVES) {
    v8f c0 = {}, c1 = {};
    #pragma unroll 2
    for (int kt = 0; kt < KSTEPS; ++kt) {
      v16bf b  = load_b_frag(w3t, nt * 16, HF, kt * 32, lane);
      v16bf a0 = load_a_frag(sH2,           HF, kt * 32, lane);
      v16bf a1 = load_a_frag(sH2 + 16 * HF, HF, kt * 32, lane);
      c0 = wmma_bf16(a0, b, c0);
      c1 = wmma_bf16(a1, b, c1);
    }
    int ncol = nt * 16 + (lane & 15);
    float bias = (ncol < 40) ? b3[ncol] : 0.f;
    #pragma unroll
    for (int i = 0; i < 8; ++i) {
      float v = c0[i] + bias;
      v = v > 0.f ? v : 0.01f * v;
      sH3[(i + hi * 8) * 48 + ncol] = v;
      float w = c1[i] + bias;
      w = w > 0.f ? w : 0.01f * w;
      sH3[(16 + i + hi * 8) * 48 + ncol] = w;
    }
  }
  __syncthreads();

  // ---- Layer 4 + sigmoid: [32x40] @ [40x1] ----
  if (tid < MT) {
    int gr = row0 + tid;
    if (gr < n) {
      float acc = b4[0];
      #pragma unroll
      for (int j = 0; j < 40; ++j) acc += sH3[tid * 48 + j] * w4[j];
      out[gr] = 1.f / (1.f + __expf(-acc));
    }
  }
}

// ---------------------------------------------------------------------------
// Driver
// ---------------------------------------------------------------------------
extern "C" void kernel_launch(void* const* d_in, const int* in_sizes, int n_in,
                              void* d_out, int out_size, void* d_ws, size_t ws_size,
                              hipStream_t stream) {
  const float* x     = (const float*)d_in[0];
  const int*   src   = (const int*)  d_in[1];
  const int*   dst   = (const int*)  d_in[2];
  const float* Wsrc1 = (const float*)d_in[3];
  const float* Wdst1 = (const float*)d_in[4];
  const float* attn1 = (const float*)d_in[5];
  const float* bias1 = (const float*)d_in[6];
  const float* res1  = (const float*)d_in[7];
  const float* Wsrc2 = (const float*)d_in[8];
  const float* Wdst2 = (const float*)d_in[9];
  const float* attn2 = (const float*)d_in[10];
  const float* bias2 = (const float*)d_in[11];
  const float* res2  = (const float*)d_in[12];
  const float* Wsrc3 = (const float*)d_in[13];
  const float* Wdst3 = (const float*)d_in[14];
  const float* attn3 = (const float*)d_in[15];
  const float* bias3 = (const float*)d_in[16];
  const float* W1    = (const float*)d_in[17];
  const float* b1    = (const float*)d_in[18];
  const float* W2    = (const float*)d_in[19];
  const float* b2    = (const float*)d_in[20];
  const float* W3    = (const float*)d_in[21];
  const float* b3    = (const float*)d_in[22];
  const float* W4    = (const float*)d_in[23];
  const float* b4    = (const float*)d_in[24];

  const int n = in_sizes[0] / 8;    // nodes
  const int e = in_sizes[1];        // edges
  float* outp = (float*)d_out;

  // Workspace carve-out (256B aligned).
  char* cur = (char*)d_ws;
  auto carve = [&](size_t bytes) {
    char* p = cur;
    cur += (bytes + 255) & ~(size_t)255;
    return p;
  };
  float*    fs    = (float*)   carve((size_t)n * HD * 4);
  float*    fd    = (float*)   carve((size_t)n * HD * 4);
  float*    rbuf  = (float*)   carve((size_t)n * HD * 4);
  float*    acc   = (float*)   carve((size_t)n * HD * 4);
  unsigned* menc  = (unsigned*)carve((size_t)n * 2 * 4);
  float*    z     = (float*)   carve((size_t)n * 2 * 4);
  float*    sbuf  = (float*)   carve((size_t)e * 2 * 4);
  float*    hatt1 = (float*)   carve((size_t)n * HD * 4);
  float*    hdef1 = (float*)   carve((size_t)n * HD * 4);
  float*    hdef2 = (float*)   carve((size_t)n * HD * 4);
  __bf16*   conc  = (__bf16*)  carve((size_t)n * 64 * 2);
  __bf16*   w1t   = (__bf16*)  carve((size_t)HF * 64 * 2);
  __bf16*   w2t   = (__bf16*)  carve((size_t)HF * HF * 2);
  __bf16*   w3t   = (__bf16*)  carve((size_t)48 * HF * 2);
  (void)ws_size; (void)n_in; (void)out_size;

  const int bProj = (n * HD + 255) / 256;
  const int bEdge = (e + 255) / 256;
  const int bEdge2 = (e * 2 + 255) / 256;

  // Weight conversion (independent of graph work).
  conv_w1<<<(HF * 64 + 255) / 256, 256, 0, stream>>>(W1, w1t);
  conv_w2<<<(HF * HF + 255) / 256, 256, 0, stream>>>(W2, w2t);
  conv_w3<<<(48 * HF + 255) / 256, 256, 0, stream>>>(W3, w3t);

  // ---- GAT layer 1 (att1) ----
  gat_proj<<<bProj, 256, 0, stream>>>(x, 8, Wsrc1, Wdst1, res1, fs, fd, rbuf, n);
  hipMemsetAsync(menc, 0, (size_t)n * 2 * 4, stream);
  hipMemsetAsync(z,    0, (size_t)n * 2 * 4, stream);
  hipMemsetAsync(acc,  0, (size_t)n * HD * 4, stream);
  gat_edge_score<<<bEdge, 256, 0, stream>>>(src, dst, fs, fd, attn1, sbuf, menc, e);
  gat_edge_exp<<<bEdge2, 256, 0, stream>>>(dst, sbuf, menc, z, e * 2);
  gat_edge_agg<<<bEdge2, 256, 0, stream>>>(src, dst, sbuf, z, fs, acc, e * 2);
  gat_finalize<<<bProj, 256, 0, stream>>>(acc, rbuf, bias1, hatt1, n);

  // ---- GAT layer 2 (def1) ----
  gat_proj<<<bProj, 256, 0, stream>>>(x, 8, Wsrc2, Wdst2, res2, fs, fd, rbuf, n);
  hipMemsetAsync(menc, 0, (size_t)n * 2 * 4, stream);
  hipMemsetAsync(z,    0, (size_t)n * 2 * 4, stream);
  hipMemsetAsync(acc,  0, (size_t)n * HD * 4, stream);
  gat_edge_score<<<bEdge, 256, 0, stream>>>(src, dst, fs, fd, attn2, sbuf, menc, e);
  gat_edge_exp<<<bEdge2, 256, 0, stream>>>(dst, sbuf, menc, z, e * 2);
  gat_edge_agg<<<bEdge2, 256, 0, stream>>>(src, dst, sbuf, z, fs, acc, e * 2);
  gat_finalize<<<bProj, 256, 0, stream>>>(acc, rbuf, bias2, hdef1, n);

  // ---- GAT layer 3 (def2): input hdef1, identity residual ----
  gat_proj<<<bProj, 256, 0, stream>>>(hdef1, 16, Wsrc3, Wdst3, nullptr, fs, fd, rbuf, n);
  hipMemsetAsync(menc, 0, (size_t)n * 2 * 4, stream);
  hipMemsetAsync(z,    0, (size_t)n * 2 * 4, stream);
  hipMemsetAsync(acc,  0, (size_t)n * HD * 4, stream);
  gat_edge_score<<<bEdge, 256, 0, stream>>>(src, dst, fs, fd, attn3, sbuf, menc, e);
  gat_edge_exp<<<bEdge2, 256, 0, stream>>>(dst, sbuf, menc, z, e * 2);
  gat_edge_agg<<<bEdge2, 256, 0, stream>>>(src, dst, sbuf, z, fs, acc, e * 2);
  gat_finalize<<<bProj, 256, 0, stream>>>(acc, hdef1, bias3, hdef2, n);

  // ---- Concatenate + fused WMMA MLP head ----
  pack_conc<<<(n * 64 + 255) / 256, 256, 0, stream>>>(hatt1, hdef2, x, conc, n);
  mlp_fused<<<(n + MT - 1) / MT, 256, 0, stream>>>(conc, w1t, b1, w2t, b2, w3t, b3,
                                                   W4, b4, outp, n);
}
